// CrossAttention_66511863546394
// MI455X (gfx1250) — compile-verified, matrix operand order checked by reading, not run on hardware
//
#include <hip/hip_runtime.h>

typedef __attribute__((ext_vector_type(16))) _Float16 v16h;
typedef __attribute__((ext_vector_type(8)))  float    v8f;
typedef __attribute__((ext_vector_type(4)))  _Float16 v4h;

#define HEADS  16
#define DHEAD  64
#define SEQ    2048
#define CH     1024
#define BATCH  4
#define MROWS  (BATCH * SEQ)   // 8192

// q/k scale = dhead^-0.25 = 64^-0.25
#define QK_SCALE 0.35355339059327373f

// ---------------------------------------------------------------------------
// gfx1250 async global->LDS copy (ASYNCcnt path). Guarded so the file still
// compiles if the builtin is absent (fallback: register-staged copy).
// Probe-confirmed param 0 type: 'int __attribute__((vector_size(16))) __device__ *'
// ---------------------------------------------------------------------------
#if defined(__AMDGCN__) && __has_builtin(__builtin_amdgcn_global_load_async_to_lds_b128)
#define HAVE_ASYNC_LDS 1
#else
#define HAVE_ASYNC_LDS 0
#endif

#if HAVE_ASYNC_LDS
typedef int i32x4v __attribute__((vector_size(16)));
typedef __attribute__((address_space(1))) i32x4v* g_i32x4_ptr;
typedef __attribute__((address_space(3))) i32x4v* l_i32x4_ptr;
#endif

__device__ __forceinline__ void async_cp16(_Float16* ldst, const _Float16* gsrc)
{
#if HAVE_ASYNC_LDS
    __builtin_amdgcn_global_load_async_to_lds_b128(
        (g_i32x4_ptr)(unsigned long long)gsrc,
        (l_i32x4_ptr)(unsigned int)(unsigned long long)ldst,
        0, 0);
#else
    *(uint4*)ldst = *(const uint4*)gsrc;
#endif
}

__device__ __forceinline__ void wait_async_copies()
{
#if HAVE_ASYNC_LDS
#if __has_builtin(__builtin_amdgcn_s_wait_asynccnt)
    __builtin_amdgcn_s_wait_asynccnt(0);
#else
    asm volatile("s_wait_asynccnt 0x0" ::: "memory");
#endif
#endif
}

union FragU { v16h h; unsigned int u[8]; };

// Gather a 16x32 f16 WMMA A/B fragment from row-major memory [rows][ld] (halves).
// Per ISA 7.12.2: lane (g*16 + r) holds row r; VGPR j (j<4) = K {2j,2j+1}+8g,
// VGPR j>=4 = K {2j+8,2j+9}+8g.  Pairs along K are contiguous -> b32/b128 loads.
__device__ __forceinline__ v16h frag_load_rm(const _Float16* src, int ld,
                                             int rowBase, int kBase, int lane)
{
    const int g = lane >> 4;
    const int r = lane & 15;
    const _Float16* p = src + (size_t)(rowBase + r) * ld + kBase + g * 8;
    FragU f;
#pragma unroll
    for (int j = 0; j < 4; ++j) {
        f.u[j]     = *(const unsigned int*)(p + 2 * j);
        f.u[j + 4] = *(const unsigned int*)(p + 16 + 2 * j);
    }
    return f.h;
}

// ---------------------------------------------------------------------------
// f32 -> f16 conversion, 4 elements/thread, grid-stride
// ---------------------------------------------------------------------------
__global__ void cvt_f32_f16(const float* __restrict__ in,
                            _Float16* __restrict__ out, int n4)
{
    int i = blockIdx.x * blockDim.x + threadIdx.x;
    int stride = gridDim.x * blockDim.x;
    for (; i < n4; i += stride) {
        float4 v = ((const float4*)in)[i];
        v4h h;
        h[0] = (_Float16)v.x; h[1] = (_Float16)v.y;
        h[2] = (_Float16)v.z; h[3] = (_Float16)v.w;
        ((v4h*)out)[i] = h;
    }
}

// ---------------------------------------------------------------------------
// Tiled WMMA GEMM:  C[M,N] = A[M,K] * B[N,K]^T
//   tile 128x128, BK=32, 256 threads (8 waves; wave = 32x64 = 2x4 fragments),
//   double-buffered LDS, async global->LDS copies, 1 barrier per K-step.
//   mode 0: out f16 [b][h][n][d] (Q/K layout), value * scale
//   mode 1: out f16 [b][h][d][n] (V transposed layout)
//   mode 2: out f32 row-major + bias   (final projection -> d_out)
// ---------------------------------------------------------------------------
__global__ __launch_bounds__(256)
void gemm_wmma(const _Float16* __restrict__ A, const _Float16* __restrict__ B,
               int M, int N, int K, int mode, float scale,
               _Float16* __restrict__ outH, float* __restrict__ outF,
               const float* __restrict__ bias)
{
    __shared__ __align__(16) _Float16 As[2][128 * 32];
    __shared__ __align__(16) _Float16 Bs[2][128 * 32];

    const int tid  = threadIdx.x;
    const int lane = tid & 31;
    const int w    = tid >> 5;          // 0..7
    const int wy   = w >> 1;            // 0..3 : 32-row band
    const int wx   = w & 1;             // 0..1 : 64-col band
    const int rowBase = blockIdx.y * 128;
    const int colBase = blockIdx.x * 128;

    v8f c[2][4] = {};

    // ---- prologue: tile 0 -> buffer 0 -------------------------------------
#if HAVE_ASYNC_LDS
#pragma unroll
    for (int t = 0; t < 2; ++t) {
        int idx = tid + t * 256;        // 0..511
        int r   = idx >> 2;             // 0..127
        int cc  = (idx & 3) * 8;        // 0,8,16,24 halves
        async_cp16(&As[0][r * 32 + cc], A + (size_t)(rowBase + r) * K + cc);
        async_cp16(&Bs[0][r * 32 + cc], B + (size_t)(colBase + r) * K + cc);
    }
    wait_async_copies();
#else
    {
        uint4 ra[2], rb[2];
#pragma unroll
        for (int t = 0; t < 2; ++t) {
            int idx = tid + t * 256;
            int r = idx >> 2, cc = (idx & 3) * 8;
            ra[t] = *(const uint4*)(A + (size_t)(rowBase + r) * K + cc);
            rb[t] = *(const uint4*)(B + (size_t)(colBase + r) * K + cc);
        }
#pragma unroll
        for (int t = 0; t < 2; ++t) {
            int idx = tid + t * 256;
            int r = idx >> 2, cc = (idx & 3) * 8;
            *(uint4*)(&As[0][r * 32 + cc]) = ra[t];
            *(uint4*)(&Bs[0][r * 32 + cc]) = rb[t];
        }
    }
#endif
    __syncthreads();

    // ---- main loop: one barrier per K-step --------------------------------
    const int nK = K / 32;
    for (int kt = 0; kt < nK; ++kt) {
        const int cur = kt & 1, nxt = cur ^ 1;
        const int k0n = (kt + 1) * 32;

#if HAVE_ASYNC_LDS
        if (k0n < K) {
#pragma unroll
            for (int t = 0; t < 2; ++t) {
                int idx = tid + t * 256;
                int r = idx >> 2, cc = (idx & 3) * 8;
                async_cp16(&As[nxt][r * 32 + cc],
                           A + (size_t)(rowBase + r) * K + k0n + cc);
                async_cp16(&Bs[nxt][r * 32 + cc],
                           B + (size_t)(colBase + r) * K + k0n + cc);
            }
        }
#else
        uint4 ra[2], rb[2];
        const bool have = (k0n < K);
        if (have) {
#pragma unroll
            for (int t = 0; t < 2; ++t) {
                int idx = tid + t * 256;
                int r = idx >> 2, cc = (idx & 3) * 8;
                ra[t] = *(const uint4*)(A + (size_t)(rowBase + r) * K + k0n + cc);
                rb[t] = *(const uint4*)(B + (size_t)(colBase + r) * K + k0n + cc);
            }
        }
#endif

        v16h a0 = frag_load_rm(As[cur], 32, 32 * wy,      0, lane);
        v16h a1 = frag_load_rm(As[cur], 32, 32 * wy + 16, 0, lane);
#pragma unroll
        for (int j = 0; j < 4; ++j) {
            v16h bf = frag_load_rm(Bs[cur], 32, 64 * wx + 16 * j, 0, lane);
            c[0][j] = __builtin_amdgcn_wmma_f32_16x16x32_f16(false, a0, false, bf,
                                                             (short)0, c[0][j], false, false);
            c[1][j] = __builtin_amdgcn_wmma_f32_16x16x32_f16(false, a1, false, bf,
                                                             (short)0, c[1][j], false, false);
        }

#if !HAVE_ASYNC_LDS
        if (have) {
#pragma unroll
            for (int t = 0; t < 2; ++t) {
                int idx = tid + t * 256;
                int r = idx >> 2, cc = (idx & 3) * 8;
                *(uint4*)(&As[nxt][r * 32 + cc]) = ra[t];
                *(uint4*)(&Bs[nxt][r * 32 + cc]) = rb[t];
            }
        }
#endif
        wait_async_copies();
        __syncthreads();
    }

    // ---- epilogue ----------------------------------------------------------
    const int g = lane >> 4, ln = lane & 15;
#pragma unroll
    for (int i = 0; i < 2; ++i) {
#pragma unroll
        for (int j = 0; j < 4; ++j) {
#pragma unroll
            for (int r = 0; r < 8; ++r) {
                int row = rowBase + 32 * wy + 16 * i + r + 8 * g;
                int col = colBase + 64 * wx + 16 * j + ln;
                float v = c[i][j][r];
                if (mode == 0) {            // Q/K: [b][h][n][d], scaled
                    int bb = row >> 11, n = row & 2047;
                    int hh = col >> 6,  d = col & 63;
                    outH[((size_t)(bb * HEADS + hh) * SEQ + n) * DHEAD + d] =
                        (_Float16)(v * scale);
                } else if (mode == 1) {     // V: [b][h][d][n]
                    int bb = row >> 11, n = row & 2047;
                    int hh = col >> 6,  d = col & 63;
                    outH[((size_t)(bb * HEADS + hh) * DHEAD + d) * SEQ + n] =
                        (_Float16)v;
                } else {                    // final: f32 + bias
                    outF[(size_t)row * N + col] = v + bias[col];
                }
            }
        }
    }
}

// ---------------------------------------------------------------------------
// Flash attention: one block per (qtile=64, head, batch). 128 threads, 4 waves.
// Wave w owns query rows 16w..16w+15.  K/V tiles double-buffered with async
// copies issued ahead of the S = Q*K^T WMMAs.
// ---------------------------------------------------------------------------
__global__ __launch_bounds__(128)
void attn_wmma(const _Float16* __restrict__ Qg, const _Float16* __restrict__ Kg,
               const _Float16* __restrict__ Vg, _Float16* __restrict__ Aout)
{
    __shared__ __align__(16) _Float16 Qs[64 * 64];
    __shared__ __align__(16) _Float16 Ks[2][64 * 64];
    __shared__ __align__(16) _Float16 Vs[2][64 * 64];  // [d][key] (V pre-transposed)
    __shared__ float    Ss[64 * 68];                   // scores f32, padded
    __shared__ _Float16 Ps[64 * 72];                   // probs f16, padded
    __shared__ float rowM[64], rowL[64], rowA[64];

    const int tid  = threadIdx.x;
    const int lane = tid & 31;
    const int w    = tid >> 5;
    const int qBase = blockIdx.x * 64;
    const int h     = blockIdx.y;
    const int b     = blockIdx.z;

    const size_t headQK = (size_t)(b * HEADS + h) * SEQ * DHEAD;
    const size_t headV  = (size_t)(b * HEADS + h) * DHEAD * SEQ;

    // ---- prologue: Q tile + key-block 0 ------------------------------------
#if HAVE_ASYNC_LDS
#pragma unroll
    for (int t = 0; t < 4; ++t) {
        int idx = tid + t * 128;          // 0..511
        int r   = idx >> 3;
        int cc  = (idx & 7) * 8;
        async_cp16(&Qs[r * 64 + cc],    Qg + headQK + (size_t)(qBase + r) * DHEAD + cc);
        async_cp16(&Ks[0][r * 64 + cc], Kg + headQK + (size_t)r * DHEAD + cc);
        async_cp16(&Vs[0][r * 64 + cc], Vg + headV + (size_t)r * SEQ + cc);
    }
    wait_async_copies();
#else
#pragma unroll
    for (int t = 0; t < 4; ++t) {
        int idx = tid + t * 128;
        int r   = idx >> 3;
        int cc  = (idx & 7) * 8;
        *(uint4*)(&Qs[r * 64 + cc]) =
            *(const uint4*)(Qg + headQK + (size_t)(qBase + r) * DHEAD + cc);
        *(uint4*)(&Ks[0][r * 64 + cc]) =
            *(const uint4*)(Kg + headQK + (size_t)r * DHEAD + cc);
        *(uint4*)(&Vs[0][r * 64 + cc]) =
            *(const uint4*)(Vg + headV + (size_t)r * SEQ + cc);
    }
#endif
    if (tid < 64) { rowM[tid] = -1e30f; rowL[tid] = 0.0f; }
    __syncthreads();

    // Q A-fragments for this wave's 16 rows (registers, reused all blocks)
    v16h qa0 = frag_load_rm(Qs, 64, 16 * w, 0,  lane);
    v16h qa1 = frag_load_rm(Qs, 64, 16 * w, 32, lane);

    v8f o[4] = {};
    const int g = lane >> 4, ln = lane & 15;

    const int NB = SEQ / 64;
    for (int kb = 0; kb < NB; ++kb) {
        const int cur = kb & 1, nxt = cur ^ 1;
        const int kBn = (kb + 1) * 64;

        // issue next key-block copies ahead of the math
#if HAVE_ASYNC_LDS
        if (kBn < SEQ) {
#pragma unroll
            for (int t = 0; t < 4; ++t) {
                int idx = tid + t * 128;
                int r   = idx >> 3;
                int cc  = (idx & 7) * 8;
                async_cp16(&Ks[nxt][r * 64 + cc],
                           Kg + headQK + (size_t)(kBn + r) * DHEAD + cc);
                async_cp16(&Vs[nxt][r * 64 + cc],
                           Vg + headV + (size_t)r * SEQ + kBn + cc);
            }
        }
#else
        uint4 rk[4], rv[4];
        const bool have = (kBn < SEQ);
        if (have) {
#pragma unroll
            for (int t = 0; t < 4; ++t) {
                int idx = tid + t * 128;
                int r   = idx >> 3;
                int cc  = (idx & 7) * 8;
                rk[t] = *(const uint4*)(Kg + headQK + (size_t)(kBn + r) * DHEAD + cc);
                rv[t] = *(const uint4*)(Vg + headV + (size_t)r * SEQ + kBn + cc);
            }
        }
#endif

        // S(16x64) = Q(16x64) * K^T : 4 column tiles, 2 K-steps each
#pragma unroll
        for (int j = 0; j < 4; ++j) {
            v8f s = {};
            v16h kb0 = frag_load_rm(Ks[cur], 64, 16 * j, 0,  lane);
            v16h kb1 = frag_load_rm(Ks[cur], 64, 16 * j, 32, lane);
            s = __builtin_amdgcn_wmma_f32_16x16x32_f16(false, qa0, false, kb0,
                                                       (short)0, s, false, false);
            s = __builtin_amdgcn_wmma_f32_16x16x32_f16(false, qa1, false, kb1,
                                                       (short)0, s, false, false);
#pragma unroll
            for (int r = 0; r < 8; ++r)
                Ss[(16 * w + r + 8 * g) * 68 + 16 * j + ln] = s[r];
        }
        __syncthreads();

        // Online softmax: one row per thread (tid < 64)
        if (tid < 64) {
            const float* srow = Ss + tid * 68;
            float mOld = rowM[tid];
            float mx = mOld;
#pragma unroll 8
            for (int j2 = 0; j2 < 64; ++j2) mx = fmaxf(mx, srow[j2]);
            _Float16* prow = Ps + tid * 72;
            float sum = 0.0f;
#pragma unroll 8
            for (int j2 = 0; j2 < 64; ++j2) {
                float p = __expf(srow[j2] - mx);
                prow[j2] = (_Float16)p;
                sum += p;
            }
            float alpha = __expf(mOld - mx);
            rowA[tid] = alpha;
            rowL[tid] = rowL[tid] * alpha + sum;
            rowM[tid] = mx;
        }
        __syncthreads();

        // Rescale O, then O += P * V
        float alph[8];
#pragma unroll
        for (int r = 0; r < 8; ++r) alph[r] = rowA[16 * w + r + 8 * g];
        v16h pa0 = frag_load_rm(Ps, 72, 16 * w, 0,  lane);
        v16h pa1 = frag_load_rm(Ps, 72, 16 * w, 32, lane);
#pragma unroll
        for (int t = 0; t < 4; ++t) {
#pragma unroll
            for (int r = 0; r < 8; ++r) o[t][r] *= alph[r];
            v16h vb0 = frag_load_rm(Vs[cur], 64, 16 * t, 0,  lane);
            v16h vb1 = frag_load_rm(Vs[cur], 64, 16 * t, 32, lane);
            o[t] = __builtin_amdgcn_wmma_f32_16x16x32_f16(false, pa0, false, vb0,
                                                          (short)0, o[t], false, false);
            o[t] = __builtin_amdgcn_wmma_f32_16x16x32_f16(false, pa1, false, vb1,
                                                          (short)0, o[t], false, false);
        }

#if !HAVE_ASYNC_LDS
        if (have) {
#pragma unroll
            for (int t = 0; t < 4; ++t) {
                int idx = tid + t * 128;
                int r   = idx >> 3;
                int cc  = (idx & 7) * 8;
                *(uint4*)(&Ks[nxt][r * 64 + cc]) = rk[t];
                *(uint4*)(&Vs[nxt][r * 64 + cc]) = rv[t];
            }
        }
#endif
        wait_async_copies();
        __syncthreads();
    }

    // ---- finalize: O /= l, write f16 A -------------------------------------
    float rinv[8];
#pragma unroll
    for (int r = 0; r < 8; ++r) rinv[r] = 1.0f / rowL[16 * w + r + 8 * g];
#pragma unroll
    for (int t = 0; t < 4; ++t) {
#pragma unroll
        for (int r = 0; r < 8; ++r) {
            int row = qBase + 16 * w + r + 8 * g;
            int col = h * DHEAD + 16 * t + ln;
            Aout[(size_t)(b * SEQ + row) * CH + col] = (_Float16)(o[t][r] * rinv[r]);
        }
    }
}

// ---------------------------------------------------------------------------
extern "C" void kernel_launch(void* const* d_in, const int* in_sizes, int n_in,
                              void* d_out, int out_size, void* d_ws, size_t ws_size,
                              hipStream_t stream)
{
    (void)in_sizes; (void)n_in; (void)out_size; (void)ws_size;
    const float* x    = (const float*)d_in[0];
    const float* Wq   = (const float*)d_in[1];
    const float* Wk   = (const float*)d_in[2];
    const float* Wv   = (const float*)d_in[3];
    const float* Wout = (const float*)d_in[4];
    const float* bout = (const float*)d_in[5];
    float* out = (float*)d_out;

    char* base = (char*)d_ws;
    _Float16* xh    = (_Float16*)(base + 0);                    // 16 MB
    _Float16* wqh   = (_Float16*)(base + 16u * 1024 * 1024);    //  2 MB
    _Float16* wkh   = (_Float16*)(base + 18u * 1024 * 1024);
    _Float16* wvh   = (_Float16*)(base + 20u * 1024 * 1024);
    _Float16* wouth = (_Float16*)(base + 22u * 1024 * 1024);
    _Float16* qws   = (_Float16*)(base + 24u * 1024 * 1024);    // 16 MB [b][h][n][d]
    _Float16* kws   = (_Float16*)(base + 40u * 1024 * 1024);    // 16 MB [b][h][n][d]
    _Float16* vws   = (_Float16*)(base + 56u * 1024 * 1024);    // 16 MB [b][h][d][n]
    _Float16* aws   = (_Float16*)(base + 72u * 1024 * 1024);    // 16 MB [bn][h*d]

    // 1) down-convert inputs to f16
    cvt_f32_f16<<<4096, 256, 0, stream>>>(x,    xh,    (MROWS * CH) / 4);
    cvt_f32_f16<<<1024, 256, 0, stream>>>(Wq,   wqh,   (CH * CH) / 4);
    cvt_f32_f16<<<1024, 256, 0, stream>>>(Wk,   wkh,   (CH * CH) / 4);
    cvt_f32_f16<<<1024, 256, 0, stream>>>(Wv,   wvh,   (CH * CH) / 4);
    cvt_f32_f16<<<1024, 256, 0, stream>>>(Wout, wouth, (CH * CH) / 4);

    // 2) QKV projections (scale folded into Q and K)
    dim3 gg(CH / 128, MROWS / 128);
    gemm_wmma<<<gg, 256, 0, stream>>>(xh, wqh, MROWS, CH, CH, 0, QK_SCALE,
                                      qws, nullptr, nullptr);
    gemm_wmma<<<gg, 256, 0, stream>>>(xh, wkh, MROWS, CH, CH, 0, QK_SCALE,
                                      kws, nullptr, nullptr);
    gemm_wmma<<<gg, 256, 0, stream>>>(xh, wvh, MROWS, CH, CH, 1, 1.0f,
                                      vws, nullptr, nullptr);

    // 3) flash attention per (qtile, head, batch)
    attn_wmma<<<dim3(SEQ / 64, HEADS, BATCH), 128, 0, stream>>>(qws, kws, vws, aws);

    // 4) output projection + bias -> f32 d_out
    gemm_wmma<<<gg, 256, 0, stream>>>(aws, wouth, MROWS, CH, CH, 2, 1.0f,
                                      nullptr, out, bout);
}
